// Attention_35613868819045
// MI455X (gfx1250) — matrix operand order repaired and checked by reading, not compiled, not run on hardware
//
#include <hip/hip_runtime.h>
#include <hip/hip_bf16.h>
#include <cstdint>

// Problem constants (match reference)
#define BB 64
#define TT 1024
#define DQ 1024
#define DV 512
#define AA 512
#define FF 32
#define KK 31

// WMMA vector types (gfx1250, wave32)
typedef __attribute__((ext_vector_type(16))) __bf16 v16bf;
typedef __attribute__((ext_vector_type(8)))  float  v8f;

union AFrag { unsigned int u[8]; v16bf v; };

__device__ __forceinline__ unsigned int f2bf(float f) {
    unsigned int x = __float_as_uint(f);
    return (x + 0x7FFFu + ((x >> 16) & 1u)) >> 16;  // RNE truncate to bf16
}

#if __has_builtin(__builtin_amdgcn_cvt_pk_bf16_f32)
typedef __attribute__((ext_vector_type(2))) __bf16 v2bf;
__device__ __forceinline__ unsigned int packbf(float lo, float hi) {
    v2bf r = __builtin_amdgcn_cvt_pk_bf16_f32(lo, hi);   // v_cvt_pk_bf16_f32
    return __builtin_bit_cast(unsigned int, r);
}
#else
__device__ __forceinline__ unsigned int packbf(float lo, float hi) {
    return f2bf(lo) | (f2bf(hi) << 16);
}
#endif

// ---------------- ws layout (byte offsets, 256B aligned) ----------------
// packB : uint[32 ntiles][17 ktiles][32 lanes][8 vgprs]      = 557056 B
// pq    : f32 [B, A]                                          = 131072 B
// convbf: bf16(ushort) [B, T, F]                              = 4194304 B
// score : f32 [B, T]                                          = 262144 B
// attnw : f32 [B, T]                                          = 262144 B
#define OFF_PACKB 0
#define OFF_PQ    557056
#define OFF_CONV  688128
#define OFF_SCORE 4882432
#define OFF_ATTNW 5144576

// ============ Kernel 1: pack W_inputs + W_loc into B-fragment order ============
// B-matrix 32x16 bf16 fragment (wave32): lane half selects K base (0 or 8);
// VGPR j<4 holds K = base+2j, j>=4 holds K = base+16+2(j-4) (lo) and +1 (hi); N = lane&15.
__global__ void pack_b_kernel(const float* __restrict__ W_inputs,
                              const float* __restrict__ W_loc,
                              unsigned int* __restrict__ packB) {
    int idx = blockIdx.x * blockDim.x + threadIdx.x;
    const int total = 32 * 17 * 32 * 8;
    if (idx >= total) return;
    int j    = idx & 7;
    int lane = (idx >> 3) & 31;
    int rem  = idx >> 8;
    int kt   = rem % 17;
    int nt   = rem / 17;
    int n      = nt * 16 + (lane & 15);
    int klocal = (lane >> 4) * 8 + 2 * (j & 3) + 16 * (j >> 2);
    float v0, v1;
    if (kt < 16) {
        int k = kt * 32 + klocal;
        v0 = W_inputs[(size_t)k * AA + n];
        v1 = W_inputs[(size_t)(k + 1) * AA + n];
    } else {  // location-feature k-tile: W_loc is [F=32, A]
        v0 = W_loc[(size_t)klocal * AA + n];
        v1 = W_loc[(size_t)(klocal + 1) * AA + n];
    }
    packB[idx] = packbf(v0, v1);
}

// ============ Kernel 2: location conv -> bf16 features [B,T,F] ============
// conv[b,t,f] = sum_k old[b,t+k-15]*lk[k,0,f] + cum[b,t+k-15]*lk[k,1,f]
__global__ void conv_loc_kernel(const float* __restrict__ attn_old,
                                const float* __restrict__ attn_cum,
                                const float* __restrict__ lk,   // [K,2,F]
                                unsigned short* __restrict__ convbf) {
    int idx = blockIdx.x * blockDim.x + threadIdx.x;  // B*T*F = 2^21
    int f = idx & (FF - 1);
    int t = (idx >> 5) & (TT - 1);
    int b = idx >> 15;
    float s = 0.f;
#pragma unroll 1
    for (int k = 0; k < KK; ++k) {
        int tp = t + k - 15;
        if (tp >= 0 && tp < TT) {
            float o = attn_old[(size_t)b * TT + tp];
            float c = attn_cum[(size_t)b * TT + tp];
            s += o * lk[(k * 2 + 0) * FF + f] + c * lk[(k * 2 + 1) * FF + f];
        }
    }
    convbf[idx] = (unsigned short)f2bf(s);
}

// ============ Kernel 3: pq = query @ W_query   [B,A] f32 ============
__global__ void pq_kernel(const float* __restrict__ query,
                          const float* __restrict__ W_query,
                          float* __restrict__ pq) {
    int idx = blockIdx.x * blockDim.x + threadIdx.x;  // B*A
    int n = idx & (AA - 1);
    int b = idx >> 9;
    float s = 0.f;
    const float* q = query + (size_t)b * DQ;
#pragma unroll 4
    for (int k = 0; k < DQ; ++k) s += q[k] * W_query[(size_t)k * AA + n];
    pq[idx] = s;
}

// ============ Kernel 4: fused score GEMM (bf16 WMMA) + tanh + v_w dot ============
// Block: (b, t-tile of 16 rows), 256 threads = 8 waves; wave w covers n-tiles
// w, w+8, w+16, w+24.  17 k-tiles: 16 from values@W_inputs + 1 from conv@W_loc.
// B fragments are double-buffered in LDS via global_load_async_to_lds_b128,
// prefetching k-tile kt+1 while WMMAs consume kt (ASYNCcnt + barrier drain).
__global__ void __launch_bounds__(256)
score_wmma_kernel(const float* __restrict__ values,          // [B,T,DV]
                  const unsigned short* __restrict__ convbf, // [B,T,F] bf16
                  const unsigned int* __restrict__ packB,    // packed B frags
                  const float* __restrict__ pq,              // [B,A]
                  const float* __restrict__ vw,              // [A]
                  const float* __restrict__ vb,              // [1]
                  float* __restrict__ score) {               // [B,T]
    const int b    = blockIdx.x;
    const int t0   = blockIdx.y * 16;
    const int tid  = threadIdx.x;
    const int lane = tid & 31;
    const int wave = tid >> 5;
    const int m     = lane & 15;       // row within half / N column index
    const int khalf = lane >> 4;       // 0: low K half, 1: high K half

    // [2 buffers][32 ntiles][32 lanes][8 uints] = 2 x 32 KB
    __shared__ __align__(16) unsigned int lds_b[2][32][32][8];
    __shared__ float lds_score[16];
    if (tid < 16) lds_score[tid] = 0.f;

    const unsigned int lds_base =
        (unsigned int)(uintptr_t)(void*)&lds_b[0][0][0][0];

    // Issue the 32 KB of B fragments for k-tile `kt` into buffer `buf`.
    // 2048 16-byte chunks; 256 threads x 8 chunks. Chunk c -> nt=c>>6,
    // rem=c&63; src uint offset (nt*17+kt)*256 + rem*4 ; dst byte c*16.
    auto issue_bt = [&](int kt, int buf) {
#pragma unroll
        for (int i = 0; i < 8; ++i) {
            int c   = i * 256 + tid;
            int nt  = c >> 6;
            int rem = c & 63;
            const unsigned int* g = packB + ((size_t)(nt * 17 + kt)) * 256 + rem * 4;
            unsigned int loff = lds_base + (unsigned int)(buf * 32768 + c * 16);
            asm volatile("global_load_async_to_lds_b128 %0, %1, off"
                         :: "v"(loff), "v"(g) : "memory");
        }
    };

    v8f acc0 = {}, acc1 = {}, acc2 = {}, acc3 = {};
    const int trow = t0 + m;
    const float* vrow = values + ((size_t)b * TT + trow) * DV;
    const unsigned short* crow = convbf + ((size_t)b * TT + trow) * FF;

    issue_bt(0, 0);  // prologue: stage k-tile 0

    for (int kt = 0; kt < 17; ++kt) {
        // Build A fragment (registers) while async B loads are in flight.
        AFrag a;
        if (kt < 16) {
            // 16-bit A 16x32 layout: lane half selects K base (0 or 8),
            // VGPR j<4 -> K = base+2j ; j>=4 -> K = base+16+2(j-4)
            const float2* src = reinterpret_cast<const float2*>(vrow + kt * 32 + khalf * 8);
#pragma unroll
            for (int j = 0; j < 4; ++j) {
                float2 p = src[j];
                a.u[j] = packbf(p.x, p.y);
            }
#pragma unroll
            for (int j = 0; j < 4; ++j) {
                float2 p = src[8 + j];
                a.u[4 + j] = packbf(p.x, p.y);
            }
        } else {
            const unsigned short* src = crow + khalf * 8;
#pragma unroll
            for (int j = 0; j < 4; ++j)
                a.u[j] = (unsigned int)src[2 * j] | ((unsigned int)src[2 * j + 1] << 16);
#pragma unroll
            for (int j = 0; j < 4; ++j)
                a.u[4 + j] = (unsigned int)src[16 + 2 * j] | ((unsigned int)src[16 + 2 * j + 1] << 16);
        }

        // Prefetch next k-tile, then drain this k-tile's 8 async loads
        // (in-order completion: asynccnt<=8 means the earlier 8 are done).
        if (kt < 16) {
            issue_bt(kt + 1, (kt + 1) & 1);
            asm volatile("s_wait_asynccnt 8" ::: "memory");
        } else {
            asm volatile("s_wait_asynccnt 0" ::: "memory");
        }
        __syncthreads();

        const int buf = kt & 1;
#pragma unroll
        for (int i = 0; i < 4; ++i) {
            int nt = wave + 8 * i;
            AFrag bf;
            const uint4* bs = reinterpret_cast<const uint4*>(&lds_b[buf][nt][lane][0]);
            uint4 q0 = bs[0];
            uint4 q1 = bs[1];
            bf.u[0] = q0.x; bf.u[1] = q0.y; bf.u[2] = q0.z; bf.u[3] = q0.w;
            bf.u[4] = q1.x; bf.u[5] = q1.y; bf.u[6] = q1.z; bf.u[7] = q1.w;
            v8f* accp = (i == 0) ? &acc0 : (i == 1) ? &acc1 : (i == 2) ? &acc2 : &acc3;
            *accp = __builtin_amdgcn_wmma_f32_16x16x32_bf16(
                false, a.v, false, bf.v, (short)0, *accp, false, false);
        }
        __syncthreads();  // all waves done reading `buf` before it is refilled
    }

    // Epilogue: C/D layout -> VGPR r: lanes 0-15 M=r, lanes 16-31 M=r+8; N=lane&15
#pragma unroll
    for (int i = 0; i < 4; ++i) {
        int nt = wave + 8 * i;
        int ncol = nt * 16 + m;
        float pqn = pq[(size_t)b * AA + ncol];
        float vwn = vw[ncol];
        v8f acc = (i == 0) ? acc0 : (i == 1) ? acc1 : (i == 2) ? acc2 : acc3;
#pragma unroll
        for (int r = 0; r < 8; ++r) {
            float v = tanhf(acc[r] + pqn) * vwn;
            // reduce over the 16 N-lanes within each half-wave
            v += __shfl_xor(v, 1);
            v += __shfl_xor(v, 2);
            v += __shfl_xor(v, 4);
            v += __shfl_xor(v, 8);
            if (m == 0) atomicAdd(&lds_score[r + khalf * 8], v);
        }
    }
    __syncthreads();
    if (tid < 16)
        score[(size_t)b * TT + t0 + tid] = lds_score[tid] + vb[0];
}

// ============ Kernel 5: softmax + forward-attention update ============
// One block per b, 256 threads x 4 elements.
__global__ void __launch_bounds__(256)
softmax_alpha_kernel(const float* __restrict__ score,  // [B,T]
                     const float* __restrict__ alpha,  // [B,T]
                     const float* __restrict__ cum,    // [B,T]
                     float* __restrict__ out_attnw,    // d_out + 32768
                     float* __restrict__ out_newcum,   // d_out + 98304
                     float* __restrict__ out_newold,   // d_out + 163840
                     float* __restrict__ out_newalpha, // d_out + 229376
                     float* __restrict__ ws_attnw) {   // ws copy for context
    const int b = blockIdx.x;
    const int tid = threadIdx.x;
    __shared__ float red[256];
    const float* s = score + (size_t)b * TT;
    const float* al = alpha + (size_t)b * TT;

    // 1) max
    float mx = -3.4e38f;
#pragma unroll
    for (int i = 0; i < 4; ++i) mx = fmaxf(mx, s[i * 256 + tid]);
    red[tid] = mx; __syncthreads();
    for (int st = 128; st > 0; st >>= 1) {
        if (tid < st) red[tid] = fmaxf(red[tid], red[tid + st]);
        __syncthreads();
    }
    mx = red[0]; __syncthreads();

    // 2) exp-sum
    float e[4]; float sum = 0.f;
#pragma unroll
    for (int i = 0; i < 4; ++i) { e[i] = __expf(s[i * 256 + tid] - mx); sum += e[i]; }
    red[tid] = sum; __syncthreads();
    for (int st = 128; st > 0; st >>= 1) {
        if (tid < st) red[tid] += red[tid + st];
        __syncthreads();
    }
    float denom = red[0]; __syncthreads();

    // 3) new_alpha (unnormalized) + sum
    float na[4]; float nasum = 0.f;
#pragma unroll
    for (int i = 0; i < 4; ++i) {
        int t = i * 256 + tid;
        float sn = e[i] / denom;
        float prev = (t > 0) ? al[t - 1] : 0.f;
        na[i] = (0.5f * al[t] + 0.5f * prev + 1e-8f) * sn;
        nasum += na[i];
    }
    red[tid] = nasum; __syncthreads();
    for (int st = 128; st > 0; st >>= 1) {
        if (tid < st) red[tid] += red[tid + st];
        __syncthreads();
    }
    nasum = red[0];

    // 4) write outputs
#pragma unroll
    for (int i = 0; i < 4; ++i) {
        int t = i * 256 + tid;
        size_t o = (size_t)b * TT + t;
        float sn = e[i] / denom;
        float w = na[i] / nasum;
        out_attnw[o]    = w;
        out_newcum[o]   = cum[o] + sn;
        out_newold[o]   = w;
        out_newalpha[o] = w;
        ws_attnw[o]     = w;
    }
}

// ============ Kernel 6: context[b,d] = sum_t w[b,t] * values[b,t,d] ============
__global__ void __launch_bounds__(128)
context_kernel(const float* __restrict__ values,  // [B,T,DV]
               const float* __restrict__ attnw,   // [B,T]
               float* __restrict__ ctx) {         // [B,DV]
    const int b = blockIdx.x;
    const int d = blockIdx.y * 128 + threadIdx.x;
    __shared__ float w[TT];
    for (int i = threadIdx.x; i < TT; i += 128) w[i] = attnw[(size_t)b * TT + i];
    __syncthreads();
    float acc = 0.f;
    const float* vb = values + (size_t)b * TT * DV + d;
#pragma unroll 4
    for (int t = 0; t < TT; ++t) acc += w[t] * vb[(size_t)t * DV];
    ctx[(size_t)b * DV + d] = acc;
}

// ============================ launch ============================
extern "C" void kernel_launch(void* const* d_in, const int* in_sizes, int n_in,
                              void* d_out, int out_size, void* d_ws, size_t ws_size,
                              hipStream_t stream) {
    const float* query     = (const float*)d_in[0];
    const float* values    = (const float*)d_in[1];
    const float* attn_cum  = (const float*)d_in[2];
    const float* attn_old  = (const float*)d_in[3];
    const float* alpha     = (const float*)d_in[4];
    const float* W_query   = (const float*)d_in[5];
    const float* W_inputs  = (const float*)d_in[6];
    const float* v_w       = (const float*)d_in[7];
    const float* v_b       = (const float*)d_in[8];
    const float* loc_kern  = (const float*)d_in[9];
    const float* W_loc     = (const float*)d_in[10];

    char* ws = (char*)d_ws;
    unsigned int*   packB  = (unsigned int*)(ws + OFF_PACKB);
    float*          pq     = (float*)(ws + OFF_PQ);
    unsigned short* convbf = (unsigned short*)(ws + OFF_CONV);
    float*          score  = (float*)(ws + OFF_SCORE);
    float*          attnw  = (float*)(ws + OFF_ATTNW);

    float* out = (float*)d_out;
    float* out_ctx      = out;                       // [B,DV]   32768
    float* out_attnw    = out + 32768;               // [B,T]    65536
    float* out_newcum   = out + 32768 + 65536;       // [B,T]
    float* out_newold   = out + 32768 + 2 * 65536;   // [B,T]
    float* out_newalpha = out + 32768 + 3 * 65536;   // [B,T]

    // 1) pack B fragments (W_inputs 16 k-tiles + W_loc 1 k-tile)
    pack_b_kernel<<<(32 * 17 * 32 * 8 + 255) / 256, 256, 0, stream>>>(
        W_inputs, W_loc, packB);
    // 2) location conv -> bf16 features
    conv_loc_kernel<<<(BB * TT * FF) / 256, 256, 0, stream>>>(
        attn_old, attn_cum, loc_kern, convbf);
    // 3) query projection
    pq_kernel<<<(BB * AA) / 256, 256, 0, stream>>>(query, W_query, pq);
    // 4) fused bf16-WMMA score (async-LDS double-buffered B)
    score_wmma_kernel<<<dim3(BB, TT / 16), 256, 0, stream>>>(
        values, convbf, packB, pq, v_w, v_b, score);
    // 5) softmax + forward attention
    softmax_alpha_kernel<<<BB, 256, 0, stream>>>(
        score, alpha, attn_cum, out_attnw, out_newcum, out_newold,
        out_newalpha, attnw);
    // 6) context reduction
    context_kernel<<<dim3(BB, DV / 128), 128, 0, stream>>>(values, attnw, out_ctx);
}
